// FlashMLAAttention_63685775065118
// MI455X (gfx1250) — matrix operand order repaired and checked by reading, not compiled, not run on hardware
//
#include <hip/hip_runtime.h>

// ---------------------------------------------------------------------------
// MHA block for MI455X (gfx1250, wave32), bf16 WMMA with f32 accumulation.
// Pipeline: qkv_proj (GEMM) -> flash_attn (online softmax) -> out_proj (GEMM)
// CDNA5 features: v_wmma_f32_16x16x32_bf16, TDM tensor_load_to_lds with LDS
// padding + double buffering (TENSORcnt), split wait counters, wave32 layout.
// Workspace usage: 4 * B*H*S*D bf16 buffers = 128 MB.
// ---------------------------------------------------------------------------

typedef __bf16 bf16;
typedef __attribute__((ext_vector_type(16))) __bf16 v16bf;
typedef __attribute__((ext_vector_type(8)))  float  v8f;
typedef __attribute__((ext_vector_type(4)))  unsigned int v4u;
typedef __attribute__((ext_vector_type(8)))  int v8i;
typedef __attribute__((ext_vector_type(4)))  int v4i;

#define NUM_HEADS 16
#define HEAD_DIM  128
#define SEQ       2048
#define EMB       2048
#define BATCH     4
#define SM_SCALE  0.5f

__device__ __forceinline__ v8f wmma_bf16(v16bf a, v16bf b, v8f c) {
  // (neg_a, A, neg_b, B, c_mod, C, reuse_a, reuse_b)
  return __builtin_amdgcn_wmma_f32_16x16x32_bf16(false, a, false, b,
                                                 (short)0, c, false, false);
}

// A-matrix fragment, 16x32 bf16 (M x K), row-major source.
// lane<16: v[0..7]=K0..7, v[8..15]=K16..23 ; lane>=16: K8..15 / K24..31.
__device__ __forceinline__ v16bf load_a16(const bf16* row, int half) {
  v16bf a;
  const bf16* p0 = row + half * 8;
  const bf16* p1 = row + 16 + half * 8;
#pragma unroll
  for (int i = 0; i < 8; ++i) a[i] = p0[i];
#pragma unroll
  for (int i = 0; i < 8; ++i) a[8 + i] = p1[i];
  return a;
}

// B-matrix fragment, 32x16 bf16 (K x N), gathered from N-major storage
// (btrow = &Bt[n][k0]): lane<16: K0..7 / K16..23 ; lane>=16: K8..15 / K24..31.
__device__ __forceinline__ v16bf load_b16(const bf16* btrow, int half) {
  v16bf b;
  const bf16* p0 = btrow + half * 8;
  const bf16* p1 = btrow + 16 + half * 8;
#pragma unroll
  for (int i = 0; i < 8; ++i) b[i] = p0[i];
#pragma unroll
  for (int i = 0; i < 8; ++i) b[8 + i] = p1[i];
  return b;
}

// ---------------------------------------------------------------------------
// TDM: 2D tile load Global -> LDS via Tensor Data Mover (D# per ISA ch.8).
// tile_d0 = contiguous elements per row, tile_d1 = rows, 2-byte elements.
// Optional LDS padding: after every 2^(pad_int_code+1) DWORDs stored, skip
// (pad_amt_code+1) DWORDs.
// ---------------------------------------------------------------------------
__device__ __forceinline__ void tdm_load_2d_bf16(
    unsigned lds_addr, unsigned long long gaddr, unsigned tile_d0,
    unsigned tile_d1, unsigned long long d0_stride, int pad_en,
    unsigned pad_int_code, unsigned pad_amt_code) {
  v4u g0;
  g0[0] = 1u;  // count=1 (valid user descriptor), gather/restore off
  g0[1] = lds_addr;
  g0[2] = (unsigned)(gaddr & 0xffffffffu);
  g0[3] = (unsigned)((gaddr >> 32) & 0x01ffffffu) | (2u << 30);  // type=2
  v8i g1;
  // data_size=1 (2B) at [17:16]; pad_enable [20]; pad_interval [24:22];
  // pad_amount [31:25]; workgroup_mask[15:0]=0 (not in a cluster).
  g1[0] = (int)((1u << 16) | ((unsigned)pad_en << 20) | (pad_int_code << 22) |
                (pad_amt_code << 25));
  // tensor dims == tile dims (no OOB clipping needed)
  g1[1] = (int)((tile_d0 & 0xffffu) << 16);                 // tensor_dim0 lo
  g1[2] = (int)(((tile_d0 >> 16) & 0xffffu) |
                ((tile_d1 & 0xffffu) << 16));               // dim0 hi | dim1 lo
  g1[3] = (int)(((tile_d1 >> 16) & 0xffffu) |
                ((tile_d0 & 0xffffu) << 16));               // dim1 hi | tile_d0
  g1[4] = (int)(tile_d1 & 0xffffu);                         // tile_d1 | tile_d2=0
  g1[5] = (int)(unsigned)(d0_stride & 0xffffffffu);         // dim0_stride lo
  g1[6] = (int)(unsigned)((d0_stride >> 32) & 0xffffu);     // hi | dim1_stride=0
  g1[7] = 0;
  v4i z4;
  z4[0] = z4[1] = z4[2] = z4[3] = 0;  // groups 2/3 unused (2D tile)
  v8i z8;
#pragma unroll
  for (int i = 0; i < 8; ++i) z8[i] = 0;
  // 6-arg toolchain variant: (g0, g1, g2, g3, extra, cpol)
  __builtin_amdgcn_tensor_load_to_lds(g0, g1, z4, z4, z8, 0);
}

__device__ __forceinline__ unsigned lds_offset_of(const void* p) {
  return (unsigned)(unsigned long long)(__SIZE_TYPE__)p;  // low 32b = LDS byte addr
}

// ---------------------------------------------------------------------------
// Kernel 1: Q/K/V projection.  y = x @ W^T + bias, stored head-split bf16.
// grid = (N/128, M/128, 3), block = 256 (8 wave32).
// ---------------------------------------------------------------------------
__global__ __launch_bounds__(256)
void qkv_proj_kernel(const float* __restrict__ x,
                     const float* __restrict__ Wq, const float* __restrict__ Wk,
                     const float* __restrict__ Wv,
                     const float* __restrict__ bq, const float* __restrict__ bk,
                     const float* __restrict__ bv,
                     bf16* __restrict__ qbuf, bf16* __restrict__ kbuf,
                     bf16* __restrict__ vbuf) {
  __shared__ bf16 lx[128][40];  // 128 rows x 32 K (pad->40, 80B rows, 16B aligned)
  __shared__ bf16 lw[128][40];

  const int tid  = threadIdx.x;
  const int lane = tid & 31, wave = tid >> 5;
  const int half = lane >> 4, ln16 = lane & 15;
  const int n0 = blockIdx.x * 128;
  const int m0 = blockIdx.y * 128;
  const int which = blockIdx.z;

  const float* W    = (which == 0) ? Wq : ((which == 1) ? Wk : Wv);
  const float* bias = (which == 0) ? bq : ((which == 1) ? bk : bv);
  bf16*        out  = (which == 0) ? qbuf : ((which == 1) ? kbuf : vbuf);

  v8f acc[8];
#pragma unroll
  for (int nf = 0; nf < 8; ++nf)
#pragma unroll
    for (int r = 0; r < 8; ++r) acc[nf][r] = 0.f;

  const int lr = tid >> 1;         // 0..127
  const int lc = (tid & 1) * 16;   // 0 / 16

  for (int kk = 0; kk < EMB; kk += 32) {
#pragma unroll
    for (int i = 0; i < 16; ++i)
      lx[lr][lc + i] = (bf16)x[(size_t)(m0 + lr) * EMB + kk + lc + i];
#pragma unroll
    for (int i = 0; i < 16; ++i)
      lw[lr][lc + i] = (bf16)W[(size_t)(n0 + lr) * EMB + kk + lc + i];
    __syncthreads();

    v16bf a = load_a16(&lx[wave * 16 + ln16][0], half);
#pragma unroll
    for (int nf = 0; nf < 8; ++nf) {
      v16bf b = load_b16(&lw[nf * 16 + ln16][0], half);
      acc[nf] = wmma_bf16(a, b, acc[nf]);
    }
    __syncthreads();
  }

  // Epilogue: bias, fold SM_SCALE into Q, head-split store [B,H,S,D].
  const float qscale = (which == 0) ? SM_SCALE : 1.0f;
#pragma unroll
  for (int nf = 0; nf < 8; ++nf) {
    const int n = n0 + nf * 16 + ln16;
    const int h = n >> 7, d = n & 127;
    const float bb = bias[n];
#pragma unroll
    for (int r = 0; r < 8; ++r) {
      const int m = m0 + wave * 16 + r + 8 * half;
      const int bidx = m >> 11, s = m & (SEQ - 1);
      float v = (acc[nf][r] + bb) * qscale;
      out[(((size_t)bidx * NUM_HEADS + h) * SEQ + s) * HEAD_DIM + d] = (bf16)v;
    }
  }
}

// ---------------------------------------------------------------------------
// Kernel 2: flash attention.  grid = (S/128, B*H), block = 256.
// K chunks streamed by TDM (double-buffered, padded LDS rows); V chunks
// transposed cooperatively; online softmax; P transposed via wave-private LDS.
// ---------------------------------------------------------------------------
__global__ __launch_bounds__(256)
void flash_attn_kernel(const bf16* __restrict__ qbuf,
                       const bf16* __restrict__ kbuf,
                       const bf16* __restrict__ vbuf,
                       bf16* __restrict__ obuf) {
  __shared__ bf16 Kl[2][32][136];   // 32 keys x 128 d, TDM-padded to 272B rows
  __shared__ bf16 Vt[2][128][40];   // transposed V: 128 d x 32 keys
  __shared__ bf16 Pl[8][16][40];    // per-wave P transpose scratch

  const int tid  = threadIdx.x;
  const int lane = tid & 31, wave = tid >> 5;
  const int half = lane >> 4, ln16 = lane & 15;
  const int bh = blockIdx.y;
  const int q0 = blockIdx.x * 128;

  const bf16* qb = qbuf + (size_t)bh * SEQ * HEAD_DIM;
  const bf16* kb = kbuf + (size_t)bh * SEQ * HEAD_DIM;
  const bf16* vb = vbuf + (size_t)bh * SEQ * HEAD_DIM;

  v16bf qa[4];
#pragma unroll
  for (int dc = 0; dc < 4; ++dc)
    qa[dc] = load_a16(qb + (size_t)(q0 + wave * 16 + ln16) * HEAD_DIM + dc * 32,
                      half);

  v8f O[8];
  float mrun[8], lrun[8];
#pragma unroll
  for (int df = 0; df < 8; ++df)
#pragma unroll
    for (int r = 0; r < 8; ++r) O[df][r] = 0.f;
#pragma unroll
  for (int r = 0; r < 8; ++r) { mrun[r] = -1e30f; lrun[r] = 0.f; }

  const int frow = tid >> 3;        // 0..31 (key row)
  const int fc0  = (tid & 7) * 16;  // 0..112 (d offset)

  // Prologue: stage chunk 0 into buffer 0.
  if (wave == 0) {
    // rows of 128 bf16 = 64 DWORDs (pad_interval code 5), pad 4 DWORDs (code 3)
    tdm_load_2d_bf16(lds_offset_of(&Kl[0][0][0]),
                     (unsigned long long)(__SIZE_TYPE__)kb,
                     HEAD_DIM, 32, HEAD_DIM, 1, 5, 3);
  }
#pragma unroll
  for (int i = 0; i < 16; ++i)
    Vt[0][fc0 + i][frow] = vb[(size_t)frow * HEAD_DIM + fc0 + i];
  if (wave == 0) __builtin_amdgcn_s_wait_tensorcnt(0);
  __syncthreads();

  for (int j = 0; j < SEQ; j += 32) {
    const int buf = (j >> 5) & 1, nxt = buf ^ 1;

    // Kick off next chunk: TDM for K, cooperative transpose-fill for V.
    if (j + 32 < SEQ) {
      if (wave == 0) {
        tdm_load_2d_bf16(lds_offset_of(&Kl[nxt][0][0]),
                         (unsigned long long)(__SIZE_TYPE__)(
                             kb + (size_t)(j + 32) * HEAD_DIM),
                         HEAD_DIM, 32, HEAD_DIM, 1, 5, 3);
      }
#pragma unroll
      for (int i = 0; i < 16; ++i)
        Vt[nxt][fc0 + i][frow] =
            vb[(size_t)(j + 32 + frow) * HEAD_DIM + fc0 + i];
      if (j + 64 < SEQ)  // global_prefetch_b8 for the chunk after next
        __builtin_prefetch(vb + (size_t)(j + 64 + frow) * HEAD_DIM + fc0, 0, 1);
    }

    // Scores: S = Q(16x128) . K_chunk^T(128x32), two 16x16 C-frags.
    v8f s0, s1;
#pragma unroll
    for (int r = 0; r < 8; ++r) { s0[r] = 0.f; s1[r] = 0.f; }
#pragma unroll
    for (int dc = 0; dc < 4; ++dc)
      s0 = wmma_bf16(qa[dc], load_b16(&Kl[buf][ln16][dc * 32], half), s0);
#pragma unroll
    for (int dc = 0; dc < 4; ++dc)
      s1 = wmma_bf16(qa[dc], load_b16(&Kl[buf][16 + ln16][dc * 32], half), s1);

    // Online softmax: rowmax / rowsum via intra-half lane reductions (wave32).
    float rescale[8];
#pragma unroll
    for (int r = 0; r < 8; ++r) {
      float pm = fmaxf(s0[r], s1[r]);
      pm = fmaxf(pm, __shfl_xor(pm, 1, 32));
      pm = fmaxf(pm, __shfl_xor(pm, 2, 32));
      pm = fmaxf(pm, __shfl_xor(pm, 4, 32));
      pm = fmaxf(pm, __shfl_xor(pm, 8, 32));
      const float mnew = fmaxf(mrun[r], pm);
      const float sc = __expf(mrun[r] - mnew);
      const float p0 = __expf(s0[r] - mnew);
      const float p1 = __expf(s1[r] - mnew);
      float ps = p0 + p1;
      ps += __shfl_xor(ps, 1, 32);
      ps += __shfl_xor(ps, 2, 32);
      ps += __shfl_xor(ps, 4, 32);
      ps += __shfl_xor(ps, 8, 32);
      lrun[r] = lrun[r] * sc + ps;
      mrun[r] = mnew;
      rescale[r] = sc;
      const int prow = r + 8 * half;            // C-layout row
      Pl[wave][prow][ln16]      = (bf16)p0;     // transpose to A-layout via LDS
      Pl[wave][prow][16 + ln16] = (bf16)p1;
    }
#pragma unroll
    for (int df = 0; df < 8; ++df)
#pragma unroll
      for (int r = 0; r < 8; ++r) O[df][r] *= rescale[r];

    // Wave-private LDS transpose: wait on split DS counter before readback.
    asm volatile("s_wait_dscnt 0x0" ::: "memory");
    v16bf pa = load_a16(&Pl[wave][ln16][0], half);

    // O += P(16x32) . V_chunk(32x128)
#pragma unroll
    for (int df = 0; df < 8; ++df) {
      v16bf bv = load_b16(&Vt[buf][df * 16 + ln16][0], half);
      O[df] = wmma_bf16(pa, bv, O[df]);
    }

    // Publish next buffers / retire current ones.
    if (wave == 0) __builtin_amdgcn_s_wait_tensorcnt(0);
    __syncthreads();
  }

  // Epilogue: normalize, merge heads into [B,S,E] bf16.
  const int bidx = bh >> 4, h = bh & 15;
#pragma unroll
  for (int r = 0; r < 8; ++r) {
    const float inv = 1.0f / lrun[r];
    const int s = q0 + wave * 16 + r + 8 * half;
#pragma unroll
    for (int df = 0; df < 8; ++df) {
      const int d = df * 16 + ln16;
      obuf[((size_t)bidx * SEQ + s) * EMB + h * HEAD_DIM + d] =
          (bf16)(O[df][r] * inv);
    }
  }
}

// ---------------------------------------------------------------------------
// Kernel 3: output projection.  out = attn @ Wo^T + bo (fp32 result).
// A-tile (bf16) streamed by TDM; fp32 weight tile convert-filled.
// grid = (N/128, M/128), block = 256.
// ---------------------------------------------------------------------------
__global__ __launch_bounds__(256)
void out_proj_kernel(const bf16* __restrict__ a, const float* __restrict__ Wo,
                     const float* __restrict__ bo, float* __restrict__ out) {
  __shared__ bf16 la[128][40];
  __shared__ bf16 lw[128][40];

  const int tid  = threadIdx.x;
  const int lane = tid & 31, wave = tid >> 5;
  const int half = lane >> 4, ln16 = lane & 15;
  const int n0 = blockIdx.x * 128;
  const int m0 = blockIdx.y * 128;

  v8f acc[8];
#pragma unroll
  for (int nf = 0; nf < 8; ++nf)
#pragma unroll
    for (int r = 0; r < 8; ++r) acc[nf][r] = 0.f;

  const int lr = tid >> 1;
  const int lc = (tid & 1) * 16;

  for (int kk = 0; kk < EMB; kk += 32) {
    if (wave == 0) {
      // A tile: 128 rows x 32 bf16, row stride EMB; rows = 16 DWORDs
      // (pad_interval code 3), pad 4 DWORDs (code 3) -> 80B LDS rows.
      tdm_load_2d_bf16(lds_offset_of(&la[0][0]),
                       (unsigned long long)(__SIZE_TYPE__)(
                           a + (size_t)m0 * EMB + kk),
                       32, 128, EMB, 1, 3, 3);
    }
#pragma unroll
    for (int i = 0; i < 16; ++i)
      lw[lr][lc + i] = (bf16)Wo[(size_t)(n0 + lr) * EMB + kk + lc + i];
    if (wave == 0) __builtin_amdgcn_s_wait_tensorcnt(0);
    __syncthreads();

    v16bf af = load_a16(&la[wave * 16 + ln16][0], half);
#pragma unroll
    for (int nf = 0; nf < 8; ++nf) {
      v16bf b = load_b16(&lw[nf * 16 + ln16][0], half);
      acc[nf] = wmma_bf16(af, b, acc[nf]);
    }
    __syncthreads();
  }

#pragma unroll
  for (int nf = 0; nf < 8; ++nf) {
    const int n = n0 + nf * 16 + ln16;
    const float bb = bo[n];
#pragma unroll
    for (int r = 0; r < 8; ++r) {
      const int m = m0 + wave * 16 + r + 8 * half;
      out[(size_t)m * EMB + n] = acc[nf][r] + bb;
    }
  }
}

// ---------------------------------------------------------------------------
extern "C" void kernel_launch(void* const* d_in, const int* in_sizes, int n_in,
                              void* d_out, int out_size, void* d_ws,
                              size_t ws_size, hipStream_t stream) {
  const float* x  = (const float*)d_in[0];
  const float* Wq = (const float*)d_in[1];
  const float* bq = (const float*)d_in[2];
  const float* Wk = (const float*)d_in[3];
  const float* bk = (const float*)d_in[4];
  const float* Wv = (const float*)d_in[5];
  const float* bv = (const float*)d_in[6];
  const float* Wo = (const float*)d_in[7];
  const float* bo = (const float*)d_in[8];

  const size_t per = (size_t)BATCH * NUM_HEADS * SEQ * HEAD_DIM;  // 16.7M elems
  bf16* qbuf = (bf16*)d_ws;
  bf16* kbuf = qbuf + per;
  bf16* vbuf = kbuf + per;
  bf16* obuf = vbuf + per;   // total 128 MB of workspace

  qkv_proj_kernel<<<dim3(EMB / 128, (BATCH * SEQ) / 128, 3), 256, 0, stream>>>(
      x, Wq, Wk, Wv, bq, bk, bv, qbuf, kbuf, vbuf);
  flash_attn_kernel<<<dim3(SEQ / 128, BATCH * NUM_HEADS), 256, 0, stream>>>(
      qbuf, kbuf, vbuf, obuf);
  out_proj_kernel<<<dim3(EMB / 128, (BATCH * SEQ) / 128), 256, 0, stream>>>(
      obuf, Wo, bo, (float*)d_out);
}
